// RefineVitBlock_24644522344930
// MI455X (gfx1250) — compile-verified, matrix operand order checked by reading, not run on hardware
//
#include <hip/hip_runtime.h>

typedef __attribute__((ext_vector_type(16))) _Float16 v16h;
typedef __attribute__((ext_vector_type(8)))  _Float16 v8h;
typedef __attribute__((ext_vector_type(8)))  float    v8f;

#define B_      2
#define C_      256
#define C2_     512
#define H_      512
#define W_      512
#define NWIN_   4096
#define NWF_    1228
#define TOK_    64

// padded LDS row strides (f16 elems): +8 keeps 16B alignment, stride%32dw==4 -> conflict-free
#define LD_XN   264     // 256 + 8
#define LD_XF   520     // 512 + 8   (as f32 staging: 260 floats/row)
#define LD_H    72      // 64 + 8

#define WMMA(a,b,c) __builtin_amdgcn_wmma_f32_16x16x32_f16(false,(a),false,(b),(short)0,(c),false,false)

// ---------------- fragment loaders ----------------
// A/B fragment from a row-major f16 matrix in LDS (rows = M for A, = N for B).
// Lane l<16: k offsets {0..7}U{16..23}; lane>=16: {8..15}U{24..31}.  Two ds_load_b128.
__device__ __forceinline__ v16h ld_frag(const _Float16* base, int ld, int row, int k0, int lane) {
  const _Float16* p = base + row * ld + k0 + ((lane & 16) ? 8 : 0);
  v8h lo = *(const v8h*)p;
  v8h hi = *(const v8h*)(p + 16);
  v16h r;
#pragma unroll
  for (int i = 0; i < 8; ++i) { r[i] = lo[i]; r[i + 8] = hi[i]; }
  return r;
}

// B fragment from packed global weights: per (n-tile,k-tile), 32 lanes x 16 contiguous f16.
__device__ __forceinline__ v16h ld_pack(const _Float16* P, int Kt, int kt, int nt, int lane) {
  const _Float16* p = P + (((size_t)(nt * Kt + kt) * 32 + (size_t)lane) << 4);
  return *(const v16h*)p;
}

__device__ __forceinline__ float gelu_f(float v) {
  return 0.5f * v * (1.f + erff(v * 0.70710678118654752f));
}
__device__ __forceinline__ float hred_max(float v) {
  v = fmaxf(v, __shfl_xor(v, 1)); v = fmaxf(v, __shfl_xor(v, 2));
  v = fmaxf(v, __shfl_xor(v, 4)); v = fmaxf(v, __shfl_xor(v, 8));
  return v;
}
__device__ __forceinline__ float hred_sum(float v) {
  v += __shfl_xor(v, 1); v += __shfl_xor(v, 2);
  v += __shfl_xor(v, 4); v += __shfl_xor(v, 8);
  return v;
}

// ---------------- small kernels ----------------
__global__ void copy_kernel(const float4* __restrict__ in, float4* __restrict__ out, size_t n) {
  size_t i = (size_t)blockIdx.x * blockDim.x + threadIdx.x;
  size_t stride = (size_t)gridDim.x * blockDim.x;
  for (; i < n; i += stride) out[i] = in[i];
}

__global__ void score_kernel(const float* __restrict__ unc, float* __restrict__ scores) {
  int i = blockIdx.x * blockDim.x + threadIdx.x;
  if (i >= B_ * NWIN_) return;
  int b = i >> 12, w = i & (NWIN_ - 1);
  const float* up = unc + (((size_t)b * H_) + (w >> 6) * 8) * W_ + (w & 63) * 8;
  float s = 0.f;
#pragma unroll
  for (int dy = 0; dy < 8; ++dy)
#pragma unroll
    for (int dx = 0; dx < 8; ++dx) s += up[dy * W_ + dx];
  scores[i] = s * (1.f / 64.f);
}

// Deterministic top-k by exact rank (ties broken by lower index, matching lax.top_k)
__global__ void select_kernel(const float* __restrict__ scores, int* __restrict__ sel) {
  int i = blockIdx.x * blockDim.x + threadIdx.x;
  if (i >= B_ * NWIN_) return;
  int b = i >> 12, w = i & (NWIN_ - 1);
  const float* sc = scores + (size_t)b * NWIN_;
  float si = sc[w];
  int rank = 0;
  for (int j = 0; j < NWIN_; ++j) {
    float sj = sc[j];
    rank += (sj > si) || (sj == si && j < w);
  }
  if (rank < NWF_) sel[b * NWF_ + rank] = w;
}

// f32 (KxN row-major) -> packed f16 WMMA-B tiles
__global__ void pack_w_kernel(const float* __restrict__ W, _Float16* __restrict__ P, int K, int N) {
  int idx = blockIdx.x * blockDim.x + threadIdx.x;
  if (idx >= K * N) return;
  int e = idx & 15, lane = (idx >> 4) & 31, tile = idx >> 9;
  int Kt = K >> 5;
  int kt = tile % Kt, nt = tile / Kt;
  int n = nt * 16 + (lane & 15);
  int k = kt * 32 + ((lane & 16) ? 8 : 0) + (e >> 3) * 16 + (e & 7);
  P[idx] = (_Float16)W[(size_t)k * N + n];
}

// ---------------- attention stage (2 heads in flight, 4 pairs) ----------------
__device__ __forceinline__ void attn_stage(
    _Float16* xf, const _Float16* pk_qkv,
    _Float16* s_q, _Float16* s_k, _Float16* s_vt, _Float16* s_s,
    int wave, int lane)
{
  const int slot = wave >> 2;          // which head of the pair this wave works on
  const int mts  = wave & 3;           // 16-row strip
  const int l15  = lane & 15;
  const int hi16 = (lane & 16) ? 8 : 0;
  const int SLOT = 64 * LD_H;          // per-slot q/k/vt size
  float obuf[16][8];                   // O held in registers until all QKV reads of xf done

#pragma unroll
  for (int hp = 0; hp < 4; ++hp) {
    __syncthreads();                   // prior pair finished reading s_q/s_k/s_vt
    // QKV GEMM for heads 2hp, 2hp+1: 96 (mt,dt,mat,slot) tile jobs over 8 waves
    for (int job = wave; job < 96; job += 8) {
      int mt = job & 3;
      int dt = (job >> 2) & 3;
      int m  = (job >> 4) % 3;         // 0=Q 1=K 2=V
      int sl = (job >> 4) / 3;
      int h  = hp * 2 + sl;
      int ntg = m * 32 + h * 4 + dt;   // n-tile inside the 1536-wide qkv weight
      v8f acc = {};
#pragma unroll
      for (int kt = 0; kt < 16; ++kt) {
        v16h a  = ld_frag(xf, LD_XF, mt * 16 + l15, kt * 32, lane);
        v16h bf = ld_pack(pk_qkv, 16, kt, ntg, lane);
        acc = WMMA(a, bf, acc);
      }
      int dcol  = dt * 16 + l15;
      int rbase = mt * 16 + hi16;
      if (m == 2) {                    // store V transposed: [dim][token]
#pragma unroll
        for (int r = 0; r < 8; ++r)
          s_vt[sl * SLOT + dcol * LD_H + rbase + r] = (_Float16)acc[r];
      } else {
        _Float16* dst = (m == 0 ? s_q : s_k) + sl * SLOT;
#pragma unroll
        for (int r = 0; r < 8; ++r)
          dst[(rbase + r) * LD_H + dcol] = (_Float16)acc[r];
      }
    }
    __syncthreads();

    // S = scale * Q K^T for this wave's (head, row-strip)
    const _Float16* q  = s_q  + slot * SLOT;
    const _Float16* kk = s_k  + slot * SLOT;
    const _Float16* vt = s_vt + slot * SLOT;
    v8f sacc[4];
#pragma unroll
    for (int nt = 0; nt < 4; ++nt) {
      v8f acc = {};
#pragma unroll
      for (int kt = 0; kt < 2; ++kt) {
        v16h a = ld_frag(q,  LD_H, mts * 16 + l15, kt * 32, lane);
        v16h b = ld_frag(kk, LD_H, nt  * 16 + l15, kt * 32, lane);
        acc = WMMA(a, b, acc);
      }
      sacc[nt] = acc;
    }
    // row softmax: each row lives in one VGPR across a 16-lane half -> shfl_xor reduce
    const float scale = 0.17677669529663687f;   // (C/HEADS)^-0.5 = 1/sqrt(32)
#pragma unroll
    for (int r = 0; r < 8; ++r) {
      float mx = -3.0e38f;
#pragma unroll
      for (int nt = 0; nt < 4; ++nt) { sacc[nt][r] *= scale; mx = fmaxf(mx, sacc[nt][r]); }
      mx = hred_max(mx);
      float sm = 0.f;
#pragma unroll
      for (int nt = 0; nt < 4; ++nt) { float e = __expf(sacc[nt][r] - mx); sacc[nt][r] = e; sm += e; }
      sm = hred_sum(sm);
      float inv = 1.f / sm;
#pragma unroll
      for (int nt = 0; nt < 4; ++nt) sacc[nt][r] *= inv;
    }
    _Float16* ss = s_s + wave * (16 * LD_H);    // this wave's 16x64 strip
#pragma unroll
    for (int nt = 0; nt < 4; ++nt)
#pragma unroll
      for (int r = 0; r < 8; ++r)
        ss[(hi16 + r) * LD_H + nt * 16 + l15] = (_Float16)sacc[nt][r];

    // O = S @ V (keep in registers)
#pragma unroll
    for (int ntd = 0; ntd < 4; ++ntd) {
      v8f acc = {};
#pragma unroll
      for (int kt = 0; kt < 2; ++kt) {
        v16h a = ld_frag(ss, LD_H, l15, kt * 32, lane);
        v16h b = ld_frag(vt, LD_H, ntd * 16 + l15, kt * 32, lane);
        acc = WMMA(a, b, acc);
      }
#pragma unroll
      for (int r = 0; r < 8; ++r) obuf[hp * 4 + ntd][r] = acc[r];
    }
  }
  __syncthreads();
  // in-place residual: xf += O (each (row,col) owned by exactly one wave)
  const int rbase = mts * 16 + hi16;
#pragma unroll
  for (int hp = 0; hp < 4; ++hp)
#pragma unroll
    for (int ntd = 0; ntd < 4; ++ntd) {
      int col = (hp * 2 + slot) * 64 + ntd * 16 + l15;
#pragma unroll
      for (int r = 0; r < 8; ++r) {
        int row = rbase + r;
        float v = (float)xf[row * LD_XF + col] + obuf[hp * 4 + ntd][r];
        xf[row * LD_XF + col] = (_Float16)v;
      }
    }
  __syncthreads();
}

// ---------------- fused per-window transformer ----------------
__global__ void __launch_bounds__(256) fused_window_kernel(
    const float* __restrict__ fm,
    const float* __restrict__ norm_w, const float* __restrict__ norm_b,
    const float* __restrict__ lin0_b, const float* __restrict__ lin1_b,
    const float* __restrict__ proj_b,
    const _Float16* __restrict__ pk_lin0, const _Float16* __restrict__ pk_qkv0,
    const _Float16* __restrict__ pk_lin1, const _Float16* __restrict__ pk_qkv1,
    const _Float16* __restrict__ pk_proj,
    const int* __restrict__ sel, float* __restrict__ out)
{
  __shared__ __align__(16) _Float16 s_xn[TOK_ * LD_XN];     // 33.0 KB  (LN output, kept for final residual)
  __shared__ __align__(16) _Float16 s_xf[TOK_ * LD_XF];     // 65.0 KB  (activations; f32 gather/output staging)
  __shared__ __align__(16) _Float16 s_q [2 * 64 * LD_H];    // 18.0 KB
  __shared__ __align__(16) _Float16 s_k [2 * 64 * LD_H];    // 18.0 KB
  __shared__ __align__(16) _Float16 s_vt[2 * 64 * LD_H];    // 18.0 KB
  __shared__ __align__(16) _Float16 s_s [8 * 16 * LD_H];    // 18.0 KB   -> 170 KB total (< 320 KB WGP)

  const int tid  = threadIdx.x;
  const int wave = tid >> 5, lane = tid & 31;
  const int l15  = lane & 15, hi16 = (lane & 16) ? 8 : 0;

  const int b   = blockIdx.x / NWF_;
  const int win = sel[blockIdx.x] & (NWIN_ - 1);
  const int y0  = (win >> 6) * 8, x0 = (win & 63) * 8;

  // warm the qkv weight path in L2 (global_prefetch)
  __builtin_prefetch(pk_qkv0 + (size_t)tid * 64, 0, 1);

  // ---- gather window (fp32 staging over s_xf; lanes walk x -> coalesced 32B segments) ----
  float* xw = (float*)s_xf;            // [token][chan], row stride 260 floats
  for (int i = tid; i < TOK_ * C_; i += 256) {
    int t = i & 63, c = i >> 6;
    xw[t * 260 + c] =
        fm[((((size_t)b * C_) + c) * H_ + (y0 + (t >> 3))) * W_ + (x0 + (t & 7))];
  }
  __syncthreads();

  // ---- layernorm -> s_xn (f16), 4 lanes per token ----
  {
    int t = tid >> 2, qr = tid & 3;
    const float* xr = xw + t * 260;
    float s = 0.f, s2 = 0.f;
    for (int c = qr * 64; c < qr * 64 + 64; ++c) { float v = xr[c]; s += v; s2 += v * v; }
    s  += __shfl_xor(s, 1);  s  += __shfl_xor(s, 2);
    s2 += __shfl_xor(s2, 1); s2 += __shfl_xor(s2, 2);
    float mu = s * (1.f / 256.f);
    float rs = rsqrtf(s2 * (1.f / 256.f) - mu * mu + 1e-5f);
    for (int c = qr * 64; c < qr * 64 + 64; ++c)
      s_xn[t * LD_XN + c] = (_Float16)((xr[c] - mu) * rs * norm_w[c] + norm_b[c]);
  }
  __syncthreads();

  // ---- lin0 + GELU: s_xf = gelu(s_xn @ W0 + b0)  (64x256 @ 256x512) ----
  for (int job = wave; job < 4 * 32; job += 8) {
    int mt = job & 3, nt = job >> 2;
    v8f acc = {};
#pragma unroll
    for (int kt = 0; kt < 8; ++kt) {
      v16h a  = ld_frag(s_xn, LD_XN, mt * 16 + l15, kt * 32, lane);
      v16h bf = ld_pack(pk_lin0, 8, kt, nt, lane);
      acc = WMMA(a, bf, acc);
    }
    int col = nt * 16 + l15;
    float bb = lin0_b[col];
    int rbase = mt * 16 + hi16;
#pragma unroll
    for (int r = 0; r < 8; ++r)
      s_xf[(rbase + r) * LD_XF + col] = (_Float16)gelu_f(acc[r] + bb);
  }
  __syncthreads();

  attn_stage(s_xf, pk_qkv0, s_q, s_k, s_vt, s_s, wave, lane);

  // ---- lin1: xf += gelu(xf @ W1 + b1), register-staged then in-place ----
  {
    float accbuf[16][8];
#pragma unroll
    for (int jj = 0; jj < 16; ++jj) {
      int job = wave + jj * 8;
      int mt = job & 3, nt = job >> 2;
      v8f acc = {};
#pragma unroll
      for (int kt = 0; kt < 16; ++kt) {
        v16h a  = ld_frag(s_xf, LD_XF, mt * 16 + l15, kt * 32, lane);
        v16h bf = ld_pack(pk_lin1, 16, kt, nt, lane);
        acc = WMMA(a, bf, acc);
      }
#pragma unroll
      for (int r = 0; r < 8; ++r) accbuf[jj][r] = acc[r];
    }
    __syncthreads();
#pragma unroll
    for (int jj = 0; jj < 16; ++jj) {
      int job = wave + jj * 8;
      int mt = job & 3, nt = job >> 2;
      int col = nt * 16 + l15;
      float bb = lin1_b[col];
      int rbase = mt * 16 + hi16;
#pragma unroll
      for (int r = 0; r < 8; ++r) {
        int row = rbase + r;
        float v = (float)s_xf[row * LD_XF + col] + gelu_f(accbuf[jj][r] + bb);
        s_xf[row * LD_XF + col] = (_Float16)v;
      }
    }
    __syncthreads();
  }

  attn_stage(s_xf, pk_qkv1, s_q, s_k, s_vt, s_s, wave, lane);

  // ---- proj + GELU + xn residual: register-staged GEMM, LDS staging, coalesced writeback ----
  {
    float pacc[8][8];
#pragma unroll
    for (int jj = 0; jj < 8; ++jj) {
      int job = wave + jj * 8;             // 64 jobs: 4 mt x 16 nt
      int mt = job & 3, nt = job >> 2;
      v8f acc = {};
#pragma unroll
      for (int kt = 0; kt < 16; ++kt) {
        v16h a  = ld_frag(s_xf, LD_XF, mt * 16 + l15, kt * 32, lane);
        v16h bf = ld_pack(pk_proj, 16, kt, nt, lane);
        acc = WMMA(a, bf, acc);
      }
#pragma unroll
      for (int r = 0; r < 8; ++r) pacc[jj][r] = acc[r];
    }
    __syncthreads();                       // all reads of s_xf complete
    float* stage = (float*)s_xf;           // [chan][token], ld 65 dwords -> conflict-free
#pragma unroll
    for (int jj = 0; jj < 8; ++jj) {
      int job = wave + jj * 8;
      int mt = job & 3, nt = job >> 2;
      int col = nt * 16 + l15;
      float bb = proj_b[col];
      int rbase = mt * 16 + hi16;
#pragma unroll
      for (int r = 0; r < 8; ++r) {
        int row = rbase + r;
        stage[col * 65 + row] = (float)s_xn[row * LD_XN + col] + gelu_f(pacc[jj][r] + bb);
      }
    }
    __syncthreads();
    // lanes walk x within the window -> coalesced 32B read-modify-write segments
    for (int i = tid; i < TOK_ * C_; i += 256) {
      int t = i & 63, c = i >> 6;
      size_t gi = ((((size_t)b * C_) + c) * H_ + (y0 + (t >> 3))) * W_ + (x0 + (t & 7));
      out[gi] += stage[c * 65 + t];        // unique element per window, windows disjoint
    }
  }
}

// ---------------- host ----------------
extern "C" void kernel_launch(void* const* d_in, const int* in_sizes, int n_in,
                              void* d_out, int out_size, void* d_ws, size_t ws_size,
                              hipStream_t stream) {
  (void)in_sizes; (void)n_in; (void)out_size; (void)ws_size;
  const float* fm     = (const float*)d_in[0];
  const float* unc    = (const float*)d_in[1];
  const float* norm_w = (const float*)d_in[2];
  const float* norm_b = (const float*)d_in[3];
  const float* lin0_w = (const float*)d_in[4];
  const float* lin0_b = (const float*)d_in[5];
  const float* qkv0_w = (const float*)d_in[6];
  const float* lin1_w = (const float*)d_in[7];
  const float* lin1_b = (const float*)d_in[8];
  const float* qkv1_w = (const float*)d_in[9];
  const float* proj_w = (const float*)d_in[10];
  const float* proj_b = (const float*)d_in[11];
  float* out = (float*)d_out;

  char* ws = (char*)d_ws;
  float* scores = (float*)ws;                    // 8192 f32
  int*   sel    = (int*)(ws + 32768);            // 2456 i32
  _Float16* pk  = (_Float16*)(ws + 65536);       // 4 MiB of packed f16 weights
  _Float16* pk_lin0 = pk;
  _Float16* pk_qkv0 = pk + 131072;
  _Float16* pk_lin1 = pk + 131072 + 786432;
  _Float16* pk_qkv1 = pk + 131072 + 786432 + 262144;
  _Float16* pk_proj = pk + 131072 + 786432 + 262144 + 786432;

  // 1) passthrough copy of the full feature map into the output
  size_t n4 = (size_t)B_ * C_ * H_ * W_ / 4;
  copy_kernel<<<4096, 256, 0, stream>>>((const float4*)fm, (float4*)out, n4);

  // 2) per-window uncertainty scores, 3) deterministic rank-based top-k
  score_kernel<<<(B_ * NWIN_ + 255) / 256, 256, 0, stream>>>(unc, scores);
  select_kernel<<<(B_ * NWIN_ + 255) / 256, 256, 0, stream>>>(scores, sel);

  // 4) pack weights f32 -> f16 WMMA-B tiles (L2-resident afterwards)
  pack_w_kernel<<<(C_ * C2_ + 255) / 256, 256, 0, stream>>>(lin0_w, pk_lin0, C_, C2_);
  pack_w_kernel<<<(C2_ * 3 * C2_ + 255) / 256, 256, 0, stream>>>(qkv0_w, pk_qkv0, C2_, 3 * C2_);
  pack_w_kernel<<<(C2_ * C2_ + 255) / 256, 256, 0, stream>>>(lin1_w, pk_lin1, C2_, C2_);
  pack_w_kernel<<<(C2_ * 3 * C2_ + 255) / 256, 256, 0, stream>>>(qkv1_w, pk_qkv1, C2_, 3 * C2_);
  pack_w_kernel<<<(C2_ * C_ + 255) / 256, 256, 0, stream>>>(proj_w, pk_proj, C2_, C_);

  // 5) fused per-window transformer (one block per selected window)
  fused_window_kernel<<<B_ * NWF_, 256, 0, stream>>>(
      fm, norm_w, norm_b, lin0_b, lin1_b, proj_b,
      pk_lin0, pk_qkv0, pk_lin1, pk_qkv1, pk_proj, sel, out);
}
